// SplineCoupling_55903294325383
// MI455X (gfx1250) — matrix operand order repaired and checked by reading, not compiled, not run on hardware
//
#include <hip/hip_runtime.h>
#include <hip/hip_bf16.h>

typedef __attribute__((ext_vector_type(16))) _Float16 v16h;
typedef __attribute__((ext_vector_type(8)))  float    v8f;

#define BATCH   2097152
#define NT32    (BATCH / 32)
#define TBOUND  5.0f
#define NBINS   8

__device__ __forceinline__ v8f wmma_f16(v16h a, v16h b, v8f c) {
  // D = A(16x32 f16) * B(32x16 f16) + C(16x16 f32)
  return __builtin_amdgcn_wmma_f32_16x16x32_f16(
      /*neg_a=*/false, a, /*neg_b=*/false, b,
      /*c_mod=*/(short)0, c, /*reuse_a=*/false, /*reuse_b=*/false);
}

__device__ __forceinline__ float softplus_f(float v) {
  return (v > 15.0f) ? v : log1pf(expf(v));
}

__global__ __launch_bounds__(256) void spline_coupling_wmma_kernel(
    const float* __restrict__ x,
    const float* __restrict__ W1, const float* __restrict__ b1,
    const float* __restrict__ W2, const float* __restrict__ b2,
    const float* __restrict__ W3, const float* __restrict__ b3,
    float* __restrict__ out)
{
  __shared__ float    s_w1[64], s_b1[64], s_b2[64], s_b3[32];
  __shared__ _Float16 s_h2[8][16 * 72];   // per-wave h2 staging (16 rows, stride 72: conflict-free)
  __shared__ float    s_raw[8][32 * 33];  // per-wave raw staging (32 rows, stride 33: conflict-free)

  const int tid = threadIdx.x;
  if (tid < 64) { s_w1[tid] = W1[tid]; s_b1[tid] = b1[tid]; s_b2[tid] = b2[tid]; }
  if (tid < 32) { s_b3[tid] = (tid < 25) ? b3[tid] : 0.0f; }
  __syncthreads();

  const int lane = tid & 31;
  const int wv   = tid >> 5;
  const int n    = lane & 15;   // column (B/C/D layouts) and row-in-tile (A layout)
  const int g    = lane >> 4;   // lane half

  // ---------------- Preload B fragments into registers ----------------
  // B layout (32x16, f16): lane half g, element i -> K = 8*g + i + (i>=8 ? 8 : 0); column = n.
  v16h B2f[4][2];   // layer2: 4 N-tiles x 2 K-chunks, value = W2[ncol][k]
#pragma unroll
  for (int t = 0; t < 4; ++t)
#pragma unroll
    for (int c = 0; c < 2; ++c)
#pragma unroll
      for (int i = 0; i < 16; ++i)
        B2f[t][c][i] = (_Float16)W2[(t * 16 + n) * 64 + (c * 32 + 8 * g + i + ((i >= 8) ? 8 : 0))];

  v16h B3f[2][2];   // layer3: 2 N-tiles (25 rows padded to 32) x 2 K-chunks
#pragma unroll
  for (int t = 0; t < 2; ++t)
#pragma unroll
    for (int c = 0; c < 2; ++c)
#pragma unroll
      for (int i = 0; i < 16; ++i) {
        const int r3 = t * 16 + n;
        B3f[t][c][i] = (r3 < 25) ? (_Float16)W3[r3 * 64 + (c * 32 + 8 * g + i + ((i >= 8) ? 8 : 0))]
                                 : (_Float16)0.0f;
      }

  // Per-lane W1/b1 in A-fragment order.
  // A layout (16x32, f16): row m = n; element i -> K = 8*g + i + (i>=8 ? 8 : 0); chunk adds 32.
  float w1r[32], b1r[32];
#pragma unroll
  for (int c = 0; c < 2; ++c)
#pragma unroll
    for (int i = 0; i < 16; ++i) {
      const int K = c * 32 + 8 * g + i + ((i >= 8) ? 8 : 0);
      w1r[c * 16 + i] = s_w1[K];
      b1r[c * 16 + i] = s_b1[K];
    }

  _Float16* h2buf  = &s_h2[wv][0];
  float*    rawbuf = &s_raw[wv][0];

  const int gwave  = blockIdx.x * 8 + wv;
  const int nwaves = gridDim.x * 8;

  for (int tile = gwave; tile < NT32; tile += nwaves) {
    const int    row = tile * 32 + lane;                  // this lane's own row
    const float2 xp  = ((const float2*)x)[row];
    const float  xf  = xp.x;
    const float  xv  = xp.y;
    const float  xfo = __shfl_xor(xf, 16, 32);            // other half's x_fix

    // -------- Two 16-row M-tiles through the same register-resident weights --------
#pragma unroll
    for (int u = 0; u < 2; ++u) {
      // A-fragment rows for sub-tile u are rows u*16 + n -> lane (u*16 + n)'s xf
      const float xfu = (u == g) ? xf : xfo;

      // Layer 1: h1 = relu(xfu * w1 + b1), built directly as A fragments
      v16h A[2];
#pragma unroll
      for (int c = 0; c < 2; ++c)
#pragma unroll
        for (int i = 0; i < 16; ++i)
          A[c][i] = (_Float16)__builtin_fmaxf(
              fmaf(xfu, w1r[c * 16 + i], b1r[c * 16 + i]), 0.0f);

      // Layer 2: h2 = relu(h1 @ W2^T + b2); 8 WMMAs; stage to LDS (transpose)
#pragma unroll
      for (int t = 0; t < 4; ++t) {
        const float bv = s_b2[t * 16 + n];
        v8f acc;
#pragma unroll
        for (int r = 0; r < 8; ++r) acc[r] = bv;
        acc = wmma_f16(A[0], B2f[t][0], acc);
        acc = wmma_f16(A[1], B2f[t][1], acc);
        // D layout: lane col = n, reg r -> row m = 8*g + r
#pragma unroll
        for (int r = 0; r < 8; ++r)
          h2buf[(8 * g + r) * 72 + t * 16 + n] =
              (_Float16)__builtin_fmaxf(acc[r], 0.0f);
      }

      // Rebuild A fragments for layer 3 from LDS (per-wave DS ordering)
      v16h A3[2];
#pragma unroll
      for (int c = 0; c < 2; ++c)
#pragma unroll
        for (int i = 0; i < 16; ++i) {
          const int K = c * 32 + 8 * g + i + ((i >= 8) ? 8 : 0);
          A3[c][i] = h2buf[n * 72 + K];
        }

      // Layer 3: raw = h2 @ W3^T + b3; 4 WMMAs; stage rows u*16.. to LDS
#pragma unroll
      for (int t = 0; t < 2; ++t) {
        const float bv = s_b3[t * 16 + n];
        v8f acc;
#pragma unroll
        for (int r = 0; r < 8; ++r) acc[r] = bv;
        acc = wmma_f16(A3[0], B3f[t][0], acc);
        acc = wmma_f16(A3[1], B3f[t][1], acc);
#pragma unroll
        for (int r = 0; r < 8; ++r)
          rawbuf[(u * 16 + 8 * g + r) * 33 + t * 16 + n] = acc[r];
      }
    }

    // -------- RQS spline: one row per lane, all 32 lanes active --------
    {
      const float* rp = rawbuf + lane * 33;
      float Wp[NBINS], Hp[NBINS], Dp[NBINS + 1];
#pragma unroll
      for (int j = 0; j < NBINS; ++j) { Wp[j] = rp[j]; Hp[j] = rp[NBINS + j]; }
#pragma unroll
      for (int j = 0; j <= NBINS; ++j) Dp[j] = rp[2 * NBINS + j];

      // softmax * 2*TB for widths and heights
      float mw = Wp[0], mh = Hp[0];
#pragma unroll
      for (int j = 1; j < NBINS; ++j) {
        mw = __builtin_fmaxf(mw, Wp[j]);
        mh = __builtin_fmaxf(mh, Hp[j]);
      }
      float w[NBINS], h[NBINS], sw = 0.0f, sh = 0.0f;
#pragma unroll
      for (int j = 0; j < NBINS; ++j) {
        w[j] = expf(Wp[j] - mw); sw += w[j];
        h[j] = expf(Hp[j] - mh); sh += h[j];
      }
      const float iw = (2.0f * TBOUND) / sw, ih = (2.0f * TBOUND) / sh;
#pragma unroll
      for (int j = 0; j < NBINS; ++j) { w[j] *= iw; h[j] *= ih; }

      float d[NBINS + 1];
#pragma unroll
      for (int j = 0; j <= NBINS; ++j) d[j] = softplus_f(Dp[j]) + 0.001f;

      float cw[NBINS + 1], ch[NBINS + 1];
      cw[0] = -TBOUND; ch[0] = -TBOUND;
#pragma unroll
      for (int j = 0; j < NBINS; ++j) { cw[j + 1] = cw[j] + w[j]; ch[j + 1] = ch[j] + h[j]; }

      int idx = 0;
#pragma unroll
      for (int j = 0; j < NBINS; ++j) idx += (cw[j + 1] < xv) ? 1 : 0;
      idx = (idx > NBINS - 1) ? (NBINS - 1) : idx;

      float w_k = 0, h_k = 0, x_k = 0, y_k = 0, d_k = 0, d_k1 = 0;
#pragma unroll
      for (int j = 0; j < NBINS; ++j) {
        const bool s = (j == idx);
        w_k  = s ? w[j]     : w_k;
        h_k  = s ? h[j]     : h_k;
        x_k  = s ? cw[j]    : x_k;
        y_k  = s ? ch[j]    : y_k;
        d_k  = s ? d[j]     : d_k;
        d_k1 = s ? d[j + 1] : d_k1;
      }

      const float s_k   = h_k / w_k;
      const float xi    = __builtin_fminf(__builtin_fmaxf((xv - x_k) / w_k, 0.0f), 1.0f);
      const float om    = 1.0f - xi;
      const float denom = s_k + (d_k1 + d_k - 2.0f * s_k) * xi * om;
      const float outp  = y_k + h_k * (s_k * xi * xi + d_k * xi * om) / denom;
      const float numer = s_k * s_k * (d_k1 * xi * xi + 2.0f * s_k * xi * om + d_k * om * om);
      float ld          = logf(numer) - 2.0f * logf(denom);

      const bool inside = (xv >= -TBOUND) && (xv <= TBOUND);
      const float yv    = inside ? outp : xv;
      ld                = inside ? ld : 0.0f;

      out[row * 2 + 0]     = xf;
      out[row * 2 + 1]     = yv;
      out[2 * BATCH + row] = ld;
    }
  }
}

extern "C" void kernel_launch(void* const* d_in, const int* in_sizes, int n_in,
                              void* d_out, int out_size, void* d_ws, size_t ws_size,
                              hipStream_t stream) {
  (void)in_sizes; (void)n_in; (void)d_ws; (void)ws_size; (void)out_size;
  const float* x  = (const float*)d_in[0];
  const float* W1 = (const float*)d_in[1];
  const float* b1 = (const float*)d_in[2];
  const float* W2 = (const float*)d_in[3];
  const float* b2 = (const float*)d_in[4];
  const float* W3 = (const float*)d_in[5];
  const float* b3 = (const float*)d_in[6];
  float* out = (float*)d_out;

  dim3 grid(2048), block(256);  // 8 waves/block, grid-stride over 65536 32-row tiles
  spline_coupling_wmma_kernel<<<grid, block, 0, stream>>>(x, W1, b1, W2, b2, W3, b3, out);
}
